// SwarmExpertPool_3513283248773
// MI455X (gfx1250) — matrix operand order repaired and checked by reading, not compiled, not run on hardware
//
#include <hip/hip_runtime.h>

typedef __attribute__((ext_vector_type(16))) __bf16 v16bf;
typedef __attribute__((ext_vector_type(2)))  __bf16 v2bf;
typedef __attribute__((ext_vector_type(8)))  float  v8f;

#define NE   8
#define NL   2
#define NB   8
#define NS   256
#define ND   512
#define NF   2048
#define NH   8
#define NHD  64
#define NTOK (NB * NS)
#define D3   (3 * ND)

// ---------------------------------------------------------------------------
// pack two fp32 into packed bf16 dword (RTNE) -> lets clang pick v_cvt_pk_bf16_f32
__device__ __forceinline__ unsigned pack_bf16(float a, float b) {
  union { v2bf v; unsigned u; } c;
  c.v.x = (__bf16)a;
  c.v.y = (__bf16)b;
  return c.u;
}

// ---------------------------------------------------------------------------
// Generic batched GEMM:  C[n,m] = act( sum_k A[n,k] * W[m,k] + bias[m] ) + resid
// A: fp32, row stride lda (k contiguous), converted to bf16 in the loader.
// W: either pre-converted bf16 (Wbf != null, row-major [M][K]) staged into LDS
//    with gfx1250 async-to-LDS DMA, or fp32 with arbitrary (wSM, wSK) strides.
// Batch z -> (bb = z/hdiv, hh = z%hdiv) with per-(b,h) base offsets.
__global__ __launch_bounds__(256)
void gemm_bf16_wmma(const float* __restrict__ A, long lda, long aSB, long aSH,
                    const unsigned short* __restrict__ Wbf,
                    const float* __restrict__ W, long wSM, long wSK, long wSB, long wSH,
                    float* __restrict__ C, long ldc, long cSB, long cSH,
                    const float* __restrict__ bias,
                    const float* __restrict__ resid,
                    int M, int N, int K, int hdiv, int do_relu)
{
  __shared__ __align__(16) unsigned short As[128][40];   // [rows][k] bf16, padded
  __shared__ __align__(16) unsigned short Ws[128][40];   // [cols][k] bf16, padded

  const int z  = blockIdx.z;
  const int bb = z / hdiv;
  const int hh = z % hdiv;
  A += (long)bb * aSB + (long)hh * aSH;
  if (W) W += (long)bb * wSB + (long)hh * wSH;
  const long cOff = (long)bb * cSB + (long)hh * cSH;

  const int tid  = threadIdx.x;
  const int lane = tid & 31;
  const int wid  = tid >> 5;
  const int half = lane >> 4;
  const int lr   = lane & 15;
  const int wn   = wid & 3;   // 4 waves along rows (32 rows each)
  const int wm   = wid >> 2;  // 2 waves along cols (64 cols each)

  // generic->LDS: low 32 bits of a generic LDS pointer are the LDS byte offset
  const unsigned wsBase = (unsigned)(uintptr_t)&Ws[0][0];

  v8f acc[2][4];
  v8f vzero = {0.f, 0.f, 0.f, 0.f, 0.f, 0.f, 0.f, 0.f};
  for (int t = 0; t < 2; ++t)
    for (int s = 0; s < 4; ++s) acc[t][s] = vzero;

  const int rowBlk = blockIdx.x * 128;
  const int colBlk = blockIdx.y * 128;

  for (int k0 = 0; k0 < K; k0 += 32) {
    // ---- stage A tile (fp32 -> bf16) ----
    for (int p = 0; p < 4; ++p) {
      int idx = tid + p * 256;       // 0..1023
      int r   = idx >> 3;            // tile row 0..127
      int c4  = (idx & 7) << 2;      // k offset 0..28
      float ax = 0.f, ay = 0.f, az = 0.f, aw = 0.f;
      int gr = rowBlk + r;
      if (gr < N) {
        const float* ap = A + (long)gr * lda + k0 + c4;
        float4 v = *(const float4*)ap;
        ax = v.x; ay = v.y; az = v.z; aw = v.w;
        if (k0 + 32 < K) __builtin_prefetch(ap + 32, 0, 1);
      }
      *(unsigned*)&As[r][c4]     = pack_bf16(ax, ay);
      *(unsigned*)&As[r][c4 + 2] = pack_bf16(az, aw);
    }

    // ---- stage W tile ----
    if (Wbf) {
      // bf16 weights: async DMA global -> LDS (16B per lane), tracked by ASYNCcnt
      for (int p = 0; p < 2; ++p) {
        int idx = tid + p * 256;     // 0..511
        int r   = idx >> 2;          // 0..127
        int c8  = (idx & 3) << 3;    // k offset 0,8,16,24
        int gm  = colBlk + r;
        unsigned ldsoff = wsBase + (unsigned)(r * 40 + c8) * 2;
        if (gm < M) {
          unsigned long long ga =
              (unsigned long long)(uintptr_t)(Wbf + (long)gm * K + k0 + c8);
          asm volatile("global_load_async_to_lds_b128 %0, %1, off"
                       :: "v"(ldsoff), "v"(ga) : "memory");
        } else {
          *(unsigned*)&Ws[r][c8]     = 0u;
          *(unsigned*)&Ws[r][c8 + 2] = 0u;
          *(unsigned*)&Ws[r][c8 + 4] = 0u;
          *(unsigned*)&Ws[r][c8 + 6] = 0u;
        }
      }
      asm volatile("s_wait_asynccnt 0x0" ::: "memory");
    } else {
      // fp32 W (activations acting as weights), arbitrary strides
      for (int p = 0; p < 4; ++p) {
        int idx = tid + p * 256;
        int r   = idx >> 3;
        int c4  = (idx & 7) << 2;
        float wx = 0.f, wy = 0.f, wz = 0.f, ww = 0.f;
        int gm = colBlk + r;
        if (gm < M) {
          if (wSK == 1) {
            const float* wp = W + (long)gm * wSM + k0 + c4;
            float4 v = *(const float4*)wp;
            wx = v.x; wy = v.y; wz = v.z; ww = v.w;
            if (k0 + 32 < K) __builtin_prefetch(wp + 32, 0, 1);
          } else {
            const float* wp = W + (long)gm * wSM + (long)(k0 + c4) * wSK;
            wx = wp[0]; wy = wp[wSK]; wz = wp[2 * wSK]; ww = wp[3 * wSK];
          }
        }
        *(unsigned*)&Ws[r][c4]     = pack_bf16(wx, wy);
        *(unsigned*)&Ws[r][c4 + 2] = pack_bf16(wz, ww);
      }
    }
    __syncthreads();

    // ---- build fragments per CDNA5 16-bit layouts and issue WMMA ----
    union FU { v16bf v; unsigned u[8]; };
    FU af[2];
    for (int t = 0; t < 2; ++t) {
      int row = wn * 32 + t * 16 + lr;
      for (int j = 0; j < 4; ++j) {
        af[t].u[j]     = *(const unsigned*)&As[row][half * 8 + 2 * j];       // K 0..15
        af[t].u[4 + j] = *(const unsigned*)&As[row][16 + half * 8 + 2 * j];  // K 16..31
      }
    }
    FU wf[4];
    for (int s = 0; s < 4; ++s) {
      int col = wm * 64 + s * 16 + lr;
      for (int j = 0; j < 8; ++j)
        wf[s].u[j] = *(const unsigned*)&Ws[col][(j >> 2) * 16 + half * 8 + 2 * (j & 3)];
    }
    for (int t = 0; t < 2; ++t)
      for (int s = 0; s < 4; ++s)
        acc[t][s] = __builtin_amdgcn_wmma_f32_16x16x32_bf16(
            false, af[t].v, false, wf[s].v, (short)0, acc[t][s], false, false);
    __syncthreads();
  }

  // ---- epilogue: C layout VGPR r -> rows r / 8+r, col = lane%16 ----
  for (int t = 0; t < 2; ++t) {
    int growBase = rowBlk + wn * 32 + t * 16 + half * 8;
    for (int s = 0; s < 4; ++s) {
      int gcol = colBlk + wm * 64 + s * 16 + lr;
      if (gcol >= M) continue;
      float bv = bias ? bias[gcol] : 0.f;
      for (int r = 0; r < 8; ++r) {
        int grow = growBase + r;
        if (grow >= N) continue;
        float v = acc[t][s][r] + bv;
        if (do_relu) v = fmaxf(v, 0.f);
        long off = cOff + (long)grow * ldc + gcol;
        if (resid) v += resid[off];
        C[off] = v;
      }
    }
  }
}

// ---------------------------------------------------------------------------
// fp32 -> bf16 weight staging (n multiple of 4)
__global__ __launch_bounds__(256)
void cvt_f32_bf16(const float* __restrict__ in, unsigned short* __restrict__ out, long n)
{
  long i = ((long)blockIdx.x * 256 + threadIdx.x) * 4;
  if (i < n) {
    float4 v = *(const float4*)(in + i);
    *(unsigned*)(out + i)     = pack_bf16(v.x, v.y);
    *(unsigned*)(out + i + 2) = pack_bf16(v.z, v.w);
  }
}

// ---------------------------------------------------------------------------
// LayerNorm over rows of width ND. One wave32 per row.
__global__ __launch_bounds__(256)
void ln_rows(const float* __restrict__ in, const float* __restrict__ w,
             const float* __restrict__ b, float* __restrict__ out, int rows)
{
  int wid = threadIdx.x >> 5, lane = threadIdx.x & 31;
  int row = blockIdx.x * 8 + wid;
  if (row >= rows) return;
  const float* x = in + (long)row * ND;
  float v[16], s = 0.f, sq = 0.f;
  for (int i = 0; i < 16; ++i) {
    float t = x[lane + i * 32];
    v[i] = t; s += t; sq += t * t;
  }
  for (int o = 16; o > 0; o >>= 1) { s += __shfl_xor(s, o, 32); sq += __shfl_xor(sq, o, 32); }
  float mean = s * (1.f / ND);
  float var  = sq * (1.f / ND) - mean * mean;
  float rs   = rsqrtf(var + 1e-5f);
  float* y = out + (long)row * ND;
  for (int i = 0; i < 16; ++i) {
    int d = lane + i * 32;
    y[d] = (v[i] - mean) * rs * w[d] + b[d];
  }
}

// Final LayerNorm + expert-weighted accumulate into d_out.
__global__ __launch_bounds__(256)
void ln_final_acc(const float* __restrict__ in, const float* __restrict__ w,
                  const float* __restrict__ b, const float* __restrict__ ew,
                  int e, float* __restrict__ out)
{
  int wid = threadIdx.x >> 5, lane = threadIdx.x & 31;
  int row = blockIdx.x * 8 + wid;
  if (row >= NTOK) return;
  const float* x = in + (long)row * ND;
  float v[16], s = 0.f, sq = 0.f;
  for (int i = 0; i < 16; ++i) {
    float t = x[lane + i * 32];
    v[i] = t; s += t; sq += t * t;
  }
  for (int o = 16; o > 0; o >>= 1) { s += __shfl_xor(s, o, 32); sq += __shfl_xor(sq, o, 32); }
  float mean = s * (1.f / ND);
  float var  = sq * (1.f / ND) - mean * mean;
  float rs   = rsqrtf(var + 1e-5f);
  float wgt  = ew[(row / NS) * NE + e];
  float* y = out + (long)row * ND;
  for (int i = 0; i < 16; ++i) {
    int d = lane + i * 32;
    y[d] += wgt * ((v[i] - mean) * rs * w[d] + b[d]);
  }
}

// Row softmax (width NS) with 1/sqrt(HD) scale folded in. One wave per row.
__global__ __launch_bounds__(256)
void softmax_rows(float* __restrict__ p, int rows)
{
  int wid = threadIdx.x >> 5, lane = threadIdx.x & 31;
  int row = blockIdx.x * 8 + wid;
  if (row >= rows) return;
  float* x = p + (long)row * NS;
  float v[8], mx = -1e30f;
  for (int i = 0; i < 8; ++i) { v[i] = x[lane + i * 32] * 0.125f; mx = fmaxf(mx, v[i]); }
  for (int o = 16; o > 0; o >>= 1) mx = fmaxf(mx, __shfl_xor(mx, o, 32));
  float s = 0.f;
  for (int i = 0; i < 8; ++i) { v[i] = __expf(v[i] - mx); s += v[i]; }
  for (int o = 16; o > 0; o >>= 1) s += __shfl_xor(s, o, 32);
  float inv = 1.f / s;
  for (int i = 0; i < 8; ++i) x[lane + i * 32] = v[i] * inv;
}

__global__ void zero_f32(float* p, long n) {
  long i = (long)blockIdx.x * 256 + threadIdx.x;
  if (i < n) p[i] = 0.f;
}

// ---------------------------------------------------------------------------
static void launch_gemm(hipStream_t st,
                        const float* A, long lda, long aSB, long aSH,
                        const unsigned short* Wbf,
                        const float* W, long wSM, long wSK, long wSB, long wSH,
                        float* C, long ldc, long cSB, long cSH,
                        const float* bias, const float* resid,
                        int M, int N, int K, int batch, int hdiv, int relu)
{
  dim3 grid((N + 127) / 128, (M + 127) / 128, batch);
  gemm_bf16_wmma<<<grid, 256, 0, st>>>(A, lda, aSB, aSH, Wbf,
                                       W, wSM, wSK, wSB, wSH,
                                       C, ldc, cSB, cSH, bias, resid,
                                       M, N, K, hdiv, relu);
}

static void stage_w(hipStream_t st, const float* w, unsigned short* wbuf, long n)
{
  cvt_f32_bf16<<<(unsigned)((n / 4 + 255) / 256), 256, 0, st>>>(w, wbuf, n);
}

extern "C" void kernel_launch(void* const* d_in, const int* in_sizes, int n_in,
                              void* d_out, int out_size, void* d_ws, size_t ws_size,
                              hipStream_t stream)
{
  (void)in_sizes; (void)n_in; (void)out_size; (void)ws_size;
  const float* x    = (const float*)d_in[0];
  const float* ew   = (const float*)d_in[1];
  const float* Wqkv = (const float*)d_in[2];
  const float* bqkv = (const float*)d_in[3];
  const float* Wo   = (const float*)d_in[4];
  const float* bo   = (const float*)d_in[5];
  const float* ln1w = (const float*)d_in[6];
  const float* ln1b = (const float*)d_in[7];
  const float* ln2w = (const float*)d_in[8];
  const float* ln2b = (const float*)d_in[9];
  const float* W1   = (const float*)d_in[10];
  const float* b1   = (const float*)d_in[11];
  const float* W2   = (const float*)d_in[12];
  const float* b2   = (const float*)d_in[13];
  const float* Wp   = (const float*)d_in[14];
  const float* bp   = (const float*)d_in[15];
  const float* lnfw = (const float*)d_in[16];
  const float* lnfb = (const float*)d_in[17];
  float* out = (float*)d_out;

  // ws layout (fp32): h | tmpA | qkv | big(scores or FFN) | attn | wbuf(bf16)
  float* ws   = (float*)d_ws;
  float* h    = ws;
  float* tA   = h   + (long)NTOK * ND;
  float* qkv  = tA  + (long)NTOK * ND;
  float* big  = qkv + (long)NTOK * D3;
  float* attn = big + (long)NTOK * NF;
  unsigned short* wbuf = (unsigned short*)(attn + (long)NTOK * ND);  // 2MB staging

  long nOut = (long)NTOK * ND;
  zero_f32<<<(unsigned)((nOut + 255) / 256), 256, 0, stream>>>(out, nOut);

  for (int e = 0; e < NE; ++e) {
    hipMemcpyAsync(h, x, sizeof(float) * NTOK * ND, hipMemcpyDeviceToDevice, stream);
    for (int l = 0; l < NL; ++l) {
      long wi = (long)e * NL + l;
      // a_in = LN1(h)
      ln_rows<<<NTOK / 8, 256, 0, stream>>>(h, ln1w + wi * ND, ln1b + wi * ND, tA, NTOK);
      // qkv = a_in @ Wqkv^T + bqkv   (bf16 weight path)
      stage_w(stream, Wqkv + wi * (long)D3 * ND, wbuf, (long)D3 * ND);
      launch_gemm(stream, tA, ND, 0, 0, wbuf,
                  nullptr, 0, 1, 0, 0,
                  qkv, D3, 0, 0, bqkv + wi * D3, nullptr,
                  D3, NTOK, ND, 1, 1, 0);
      // scores[b,h] = Q @ K^T   (64 batches of 256x256x64, fp32 W path)
      launch_gemm(stream, qkv, D3, (long)NS * D3, NHD, nullptr,
                  qkv + ND, D3, 1, (long)NS * D3, NHD,
                  big, NS, (long)NH * NS * NS, (long)NS * NS,
                  nullptr, nullptr, NS, NS, NHD, NB * NH, NH, 0);
      // P = softmax(scores / 8)
      softmax_rows<<<(NB * NH * NS) / 8, 256, 0, stream>>>(big, NB * NH * NS);
      // attn[b,h] = P @ V   (W = V^T via strides, fp32 W path)
      launch_gemm(stream, big, NS, (long)NH * NS * NS, (long)NS * NS, nullptr,
                  qkv + 2 * ND, 1, D3, (long)NS * D3, NHD,
                  attn, ND, (long)NS * ND, NHD,
                  nullptr, nullptr, NHD, NS, NS, NB * NH, NH, 0);
      // h = h + attn @ Wo^T + bo
      stage_w(stream, Wo + wi * (long)ND * ND, wbuf, (long)ND * ND);
      launch_gemm(stream, attn, ND, 0, 0, wbuf,
                  nullptr, 0, 1, 0, 0,
                  h, ND, 0, 0, bo + wi * ND, h,
                  ND, NTOK, ND, 1, 1, 0);
      // f_in = LN2(h)
      ln_rows<<<NTOK / 8, 256, 0, stream>>>(h, ln2w + wi * ND, ln2b + wi * ND, tA, NTOK);
      // u = relu(f_in @ W1^T + b1)
      stage_w(stream, W1 + wi * (long)NF * ND, wbuf, (long)NF * ND);
      launch_gemm(stream, tA, ND, 0, 0, wbuf,
                  nullptr, 0, 1, 0, 0,
                  big, NF, 0, 0, b1 + wi * NF, nullptr,
                  NF, NTOK, ND, 1, 1, 1);
      // h = h + u @ W2^T + b2
      stage_w(stream, W2 + wi * (long)ND * NF, wbuf, (long)ND * NF);
      launch_gemm(stream, big, NF, 0, 0, wbuf,
                  nullptr, 0, 1, 0, 0,
                  h, ND, 0, 0, b2 + wi * ND, h,
                  ND, NTOK, NF, 1, 1, 0);
    }
    // proj: tA = h @ Wp^T + bp
    stage_w(stream, Wp + (long)e * ND * ND, wbuf, (long)ND * ND);
    launch_gemm(stream, h, ND, 0, 0, wbuf,
                nullptr, 0, 1, 0, 0,
                tA, ND, 0, 0, bp + (long)e * ND, nullptr,
                ND, NTOK, ND, 1, 1, 0);
    // out += ew[b,e] * LNf(tA)
    ln_final_acc<<<NTOK / 8, 256, 0, stream>>>(tA, lnfw + (long)e * ND,
                                               lnfb + (long)e * ND, ew, e, out);
  }
}
